// VQVAE2_38706245271709
// MI455X (gfx1250) — matrix operand order repaired and checked by reading, not compiled, not run on hardware
//
#include <hip/hip_runtime.h>
#include <hip/hip_bf16.h>

// ---------------- model dims ----------------
#define BB 16
#define TT 512
#define VV 128
#define DD 512
#define HH 8
#define LL 4
#define DFF 2048
#define KK 1024
#define DC 64
#define DH 64
#define NTOK (BB * TT)                  // 8192
#define NOUT ((size_t)NTOK * VV)        // 1048576 recon elems

typedef __bf16 v16bf __attribute__((ext_vector_type(16)));
typedef float  v8f   __attribute__((ext_vector_type(8)));

__device__ __forceinline__ __bf16 to_bf16(float f) { return (__bf16)f; }

__device__ __forceinline__ v8f wmma_bf16(v16bf a, v16bf b, v8f c) {
    return __builtin_amdgcn_wmma_f32_16x16x32_bf16(false, a, false, b, (short)0, c, false, false);
}

// Async global->LDS copy of 16 bytes (CDNA5 ASYNCcnt path, no VGPR round trip).
__device__ __forceinline__ void async_copy_b128(void* lds_dst, const void* gmem_src) {
    unsigned ldsoff = (unsigned)(unsigned long long)lds_dst;   // LDS byte offset
    asm volatile("global_load_async_to_lds_b128 %0, %1, off"
                 :: "v"(ldsoff), "v"(gmem_src) : "memory");
}
__device__ __forceinline__ void wait_async0() {
    asm volatile("s_wait_asynccnt 0x0" ::: "memory");
}

// ---------------- GEMM: C = act(A[M,K] @ W[K,N] + bias[N]) ----------------
// block = 256 threads = 8 waves in 4(M) x 2(N) grid; block tile 128 x 64;
// each wave computes a 32x32 region = 2x2 WMMA tiles with fragment reuse.
// A tile is DMA'd global->LDS via async b128 copies, then converted f32->bf16 in LDS.
template <int ACT>
__global__ __launch_bounds__(256) void gemm_kernel(const float* __restrict__ A,
                                                   const float* __restrict__ W,
                                                   const float* __restrict__ bias,
                                                   float* __restrict__ C,
                                                   int M, int N, int Kd) {
    __shared__ float  Afs[128][32];   // async-staged f32 A tile (16 KB)
    __shared__ __bf16 As[128][32];    // bf16 A tile for WMMA fragments
    __shared__ __bf16 Bs[32][64];
    const int tid  = threadIdx.x;
    const int lane = tid & 31;
    const int wid  = tid >> 5;
    const int wm   = wid & 3;         // 4 wave rows (each 32 rows of C)
    const int wn   = wid >> 2;        // 2 wave cols (each 32 cols of C)
    const int m0   = blockIdx.x * 128;
    const int n0   = blockIdx.y * 64;
    const int g    = lane >> 4;
    const int ln   = lane & 15;

    v8f acc[2][2] = {};
    for (int k0 = 0; k0 < Kd; k0 += 32) {
        // 1) async DMA the A tile into LDS: 128 rows x 32 floats = 1024 x 16B chunks
        {
            int chunk = tid;
#pragma unroll
            for (int t = 0; t < 4; ++t, chunk += 256) {
                int r = chunk >> 3;            // 8 x 16B chunks per row
                int c = (chunk & 7) << 2;      // float offset within row
                async_copy_b128(&Afs[r][c], &A[(long)(m0 + r) * Kd + k0 + c]);
            }
        }
        // 2) stage B tile (bf16 convert through VGPRs)
        for (int idx = tid; idx < 32 * 64; idx += 256) {
            int r = idx >> 6, c = idx & 63;
            Bs[r][c] = to_bf16(W[(long)(k0 + r) * N + n0 + c]);
        }
        // prefetch next K-slice of the (L2-resident) weight stream
        if (k0 + 32 < Kd)
            __builtin_prefetch(&W[(long)(k0 + 32 + (tid >> 3)) * N + n0 + ((tid & 7) << 3)], 0, 1);
        wait_async0();
        __syncthreads();
        // 3) convert A tile f32 -> bf16 entirely inside LDS
        for (int idx = tid; idx < 128 * 32; idx += 256) {
            int r = idx >> 5, c = idx & 31;
            As[r][c] = to_bf16(Afs[r][c]);
        }
        __syncthreads();
        // 4) fragments + 4 WMMAs with A/B reuse
        v16bf af[2], bf_[2];
#pragma unroll
        for (int e = 0; e < 16; ++e) {
            int ka = e + (e & 8) + 8 * g;          // A-fragment K index (ISA wave32 layout)
            af[0][e] = As[wm * 32 + ln][ka];
            af[1][e] = As[wm * 32 + 16 + ln][ka];
            int kb = e + (e & 8) + 8 * g;          // B-fragment K index
            bf_[0][e] = Bs[kb][wn * 32 + ln];
            bf_[1][e] = Bs[kb][wn * 32 + 16 + ln];
        }
#pragma unroll
        for (int i = 0; i < 2; ++i)
#pragma unroll
            for (int j = 0; j < 2; ++j)
                acc[i][j] = wmma_bf16(af[i], bf_[j], acc[i][j]);
        __syncthreads();
    }
#pragma unroll
    for (int i = 0; i < 2; ++i)
#pragma unroll
        for (int j = 0; j < 2; ++j)
#pragma unroll
            for (int r = 0; r < 8; ++r) {
                int row = m0 + wm * 32 + i * 16 + r + 8 * g;
                int col = n0 + wn * 32 + j * 16 + ln;
                float v = acc[i][j][r] + bias[col];
                if (ACT == 1) v = fmaxf(v, 0.0f);
                if (ACT == 2) v = 1.0f / (1.0f + __expf(-v));
                C[(long)row * N + col] = v;
            }
}

// ---------------- fused attention: one wave handles 16 query rows of one (b,h) ----------------
__global__ __launch_bounds__(32) void attn_kernel(const float* __restrict__ Q,
                                                  const float* __restrict__ Km,
                                                  const float* __restrict__ Vm,
                                                  float* __restrict__ O) {
    const int it = blockIdx.x;    // query tile (T/16)
    const int h  = blockIdx.y;
    const int b  = blockIdx.z;
    const int lane = threadIdx.x;
    const int g  = lane >> 4;
    const int ln = lane & 15;

    __shared__ __bf16 Qs[16][DH];
    __shared__ __bf16 Ks[16][DH];
    __shared__ float  Ss[16][TT];
    __shared__ __bf16 Ps[16][TT];
    __shared__ __bf16 Vs[32][DH];

    const long baseQ = ((long)(b * TT + it * 16)) * DD + h * DH;
    for (int idx = lane; idx < 16 * DH; idx += 32) {
        int r = idx >> 6, c = idx & 63;
        Qs[r][c] = to_bf16(Q[baseQ + (long)r * DD + c]);
    }
    __syncthreads();

    // S = Q K^T * 1/sqrt(DH)
    for (int jt = 0; jt < TT / 16; ++jt) {
        const long baseK = ((long)(b * TT + jt * 16)) * DD + h * DH;
        for (int idx = lane; idx < 16 * DH; idx += 32) {
            int r = idx >> 6, c = idx & 63;
            Ks[r][c] = to_bf16(Km[baseK + (long)r * DD + c]);
        }
        __syncthreads();
        v8f acc = {};
#pragma unroll
        for (int kk = 0; kk < 2; ++kk) {
            v16bf af, bf_;
#pragma unroll
            for (int e = 0; e < 16; ++e) {
                af[e]  = Qs[ln][kk * 32 + e + (e & 8) + 8 * g];
                bf_[e] = Ks[ln][kk * 32 + e + (e & 8) + 8 * g];   // B[k=d][n=j] = K[j][d]
            }
            acc = wmma_bf16(af, bf_, acc);
        }
#pragma unroll
        for (int r = 0; r < 8; ++r)
            Ss[r + 8 * g][jt * 16 + ln] = acc[r] * 0.125f;   // 1/sqrt(64)
        __syncthreads();
    }

    // softmax rows (fp32), emit probs as bf16
    for (int r = 0; r < 16; ++r) {
        float mx = -1e30f;
        for (int c = lane; c < TT; c += 32) mx = fmaxf(mx, Ss[r][c]);
        for (int off = 16; off > 0; off >>= 1) mx = fmaxf(mx, __shfl_xor(mx, off, 32));
        float sum = 0.0f;
        for (int c = lane; c < TT; c += 32) {
            float e = __expf(Ss[r][c] - mx);
            Ss[r][c] = e;
            sum += e;
        }
        for (int off = 16; off > 0; off >>= 1) sum += __shfl_xor(sum, off, 32);
        float inv = 1.0f / sum;
        for (int c = lane; c < TT; c += 32) Ps[r][c] = to_bf16(Ss[r][c] * inv);
    }
    __syncthreads();

    // O = P @ V
    v8f oacc[DH / 16] = {};
    for (int kt = 0; kt < TT / 32; ++kt) {
        const long baseV = ((long)(b * TT + kt * 32)) * DD + h * DH;
        for (int idx = lane; idx < 32 * DH; idx += 32) {
            int r = idx >> 6, c = idx & 63;
            Vs[r][c] = to_bf16(Vm[baseV + (long)r * DD + c]);
        }
        __syncthreads();
        v16bf af;
#pragma unroll
        for (int e = 0; e < 16; ++e)
            af[e] = Ps[ln][kt * 32 + e + (e & 8) + 8 * g];
#pragma unroll
        for (int nt = 0; nt < DH / 16; ++nt) {
            v16bf bf_;
#pragma unroll
            for (int e = 0; e < 16; ++e)
                bf_[e] = Vs[e + (e & 8) + 8 * g][nt * 16 + ln];
            oacc[nt] = wmma_bf16(af, bf_, oacc[nt]);
        }
        __syncthreads();
    }
    const long baseO = ((long)(b * TT + it * 16)) * DD + h * DH;
#pragma unroll
    for (int nt = 0; nt < DH / 16; ++nt)
#pragma unroll
        for (int r = 0; r < 8; ++r)
            O[baseO + (long)(r + 8 * g) * DD + nt * 16 + ln] = oacc[nt][r];
}

// ---------------- LayerNorm with residual: y = g*(x+r - m)/sqrt(v+eps) + b ----------------
__global__ __launch_bounds__(128) void ln_kernel(const float* __restrict__ x,
                                                 const float* __restrict__ r,
                                                 const float* __restrict__ gg,
                                                 const float* __restrict__ bb,
                                                 float* __restrict__ y) {
    const int t = blockIdx.x;
    const int tid = threadIdx.x;
    __shared__ float row[DD];
    __shared__ float red[128];
    float s = 0.0f;
    for (int c = tid; c < DD; c += 128) {
        float v = x[(long)t * DD + c] + r[(long)t * DD + c];
        row[c] = v;
        s += v;
    }
    red[tid] = s; __syncthreads();
    for (int o = 64; o > 0; o >>= 1) { if (tid < o) red[tid] += red[tid + o]; __syncthreads(); }
    float mean = red[0] / (float)DD;
    __syncthreads();
    float sq = 0.0f;
    for (int c = tid; c < DD; c += 128) { float d = row[c] - mean; sq += d * d; }
    red[tid] = sq; __syncthreads();
    for (int o = 64; o > 0; o >>= 1) { if (tid < o) red[tid] += red[tid + o]; __syncthreads(); }
    float inv = rsqrtf(red[0] / (float)DD + 1e-5f);
    for (int c = tid; c < DD; c += 128)
        y[(long)t * DD + c] = gg[c] * (row[c] - mean) * inv + bb[c];
}

// ---------------- positional encoding add (computed on the fly) ----------------
__global__ void add_pe_kernel(float* __restrict__ x) {
    long i = (long)blockIdx.x * blockDim.x + threadIdx.x;
    if (i >= (long)NTOK * DD) return;
    int d = (int)(i & (DD - 1));
    int t = (int)((i >> 9) & (TT - 1));
    int ii = d & ~1;
    float ang = (float)t * __powf(10000.0f, -(float)ii / (float)DD);
    x[i] += (d & 1) ? __cosf(ang) : __sinf(ang);
}

// ---------------- VQ: argmin over codebook, gather, accumulate MSE ----------------
__global__ __launch_bounds__(128) void vq_kernel(const float* __restrict__ z,
                                                 const float* __restrict__ cb,
                                                 float* __restrict__ zq,
                                                 float* __restrict__ mse_acc) {
    const int t = blockIdx.x;
    const int tid = threadIdx.x;
    __shared__ float zs[DC];
    __shared__ float bval[128];
    __shared__ int   bidx[128];
    if (tid < DC) zs[tid] = z[(long)t * DC + tid];
    __syncthreads();
    float best = 3.4e38f; int bi = 0;
    for (int c = tid; c < KK; c += 128) {
        const float* cc = cb + (long)c * DC;
        float d2 = 0.0f;
#pragma unroll 8
        for (int j = 0; j < DC; ++j) { float df = zs[j] - cc[j]; d2 += df * df; }
        if (d2 < best) { best = d2; bi = c; }
    }
    bval[tid] = best; bidx[tid] = bi;
    __syncthreads();
    for (int o = 64; o > 0; o >>= 1) {
        if (tid < o) {
            bool take = (bval[tid + o] < bval[tid]) ||
                        (bval[tid + o] == bval[tid] && bidx[tid + o] < bidx[tid]);
            if (take) { bval[tid] = bval[tid + o]; bidx[tid] = bidx[tid + o]; }
        }
        __syncthreads();
    }
    const int winner = bidx[0];
    __syncthreads();
    float part = 0.0f;
    if (tid < DC) {
        float cv = cb[(long)winner * DC + tid];
        zq[(long)t * DC + tid] = cv;
        float df = cv - zs[tid];
        part = df * df;
    }
    bval[tid] = part; __syncthreads();
    for (int o = 64; o > 0; o >>= 1) { if (tid < o) bval[tid] += bval[tid + o]; __syncthreads(); }
    if (tid == 0) atomicAdd(mse_acc, bval[0]);
}

// ---------------- BCE reduction ----------------
__global__ __launch_bounds__(256) void bce_kernel(const float* __restrict__ recon,
                                                  const float* __restrict__ tgt,
                                                  float* __restrict__ acc) {
    __shared__ float red[256];
    const long n = (long)NOUT;
    float s = 0.0f;
    for (long i = (long)blockIdx.x * blockDim.x + threadIdx.x; i < n;
         i += (long)gridDim.x * blockDim.x) {
        float pc = fminf(fmaxf(recon[i], 1e-7f), 1.0f - 1e-7f);
        float tv = tgt[i];
        s -= tv * __logf(pc) + (1.0f - tv) * __logf(1.0f - pc);
    }
    red[threadIdx.x] = s; __syncthreads();
    for (int o = 128; o > 0; o >>= 1) { if (threadIdx.x < o) red[threadIdx.x] += red[threadIdx.x + o]; __syncthreads(); }
    if (threadIdx.x == 0) atomicAdd(acc, red[0]);
}

__global__ void init_acc_kernel(float* acc) { acc[0] = 0.0f; acc[1] = 0.0f; }

__global__ void fin_kernel(float* __restrict__ out, const float* __restrict__ acc) {
    out[NOUT]     = acc[0] / (float)NOUT;                 // recon_loss
    out[NOUT + 1] = acc[1] / (float)((long)NTOK * DC);    // embed_loss
    out[NOUT + 2] = acc[1] / (float)((long)NTOK * DC);    // commit_loss (same forward value)
}

// ---------------- host-side param structs ----------------
struct MLP3 { const float *W1, *b1, *W2, *b2, *W3, *b3; };
struct AttnP { const float *Wq, *bq, *Wk, *bk, *Wv, *bv, *Wo, *bo; };
struct LNP { const float *g, *b; };
struct FFP { const float *W1, *b1, *W2, *b2; };
struct EncL { AttnP sa; LNP ln1; FFP ff; LNP ln2; };
struct DecL { AttnP sa; LNP ln1; AttnP ca; LNP ln2; FFP ff; LNP ln3; };

static void launch_gemm(const float* A, const float* W, const float* b, float* C,
                        int M, int N, int Kd, int act, hipStream_t s) {
    dim3 grid(M / 128, N / 64), blk(256);
    if (act == 0)      gemm_kernel<0><<<grid, blk, 0, s>>>(A, W, b, C, M, N, Kd);
    else if (act == 1) gemm_kernel<1><<<grid, blk, 0, s>>>(A, W, b, C, M, N, Kd);
    else               gemm_kernel<2><<<grid, blk, 0, s>>>(A, W, b, C, M, N, Kd);
}

struct Bufs { float *mem, *x, *qb, *kb, *vb, *t1, *ffh, *z, *zq, *acc; };

static void run_mha(const float* xq, const float* xkv, const AttnP& p, Bufs& B_, hipStream_t s) {
    launch_gemm(xq,  p.Wq, p.bq, B_.qb, NTOK, DD, DD, 0, s);
    launch_gemm(xkv, p.Wk, p.bk, B_.kb, NTOK, DD, DD, 0, s);
    launch_gemm(xkv, p.Wv, p.bv, B_.vb, NTOK, DD, DD, 0, s);
    attn_kernel<<<dim3(TT / 16, HH, BB), 32, 0, s>>>(B_.qb, B_.kb, B_.vb, B_.t1);
    launch_gemm(B_.t1, p.Wo, p.bo, B_.kb, NTOK, DD, DD, 0, s);  // result in kb
}

static void run_ff_ln(float* x, const FFP& f, const LNP& ln, Bufs& B_, hipStream_t s) {
    launch_gemm(x, f.W1, f.b1, B_.ffh, NTOK, DFF, DD, 1, s);
    launch_gemm(B_.ffh, f.W2, f.b2, B_.t1, NTOK, DD, DFF, 0, s);
    ln_kernel<<<NTOK, 128, 0, s>>>(x, B_.t1, ln.g, ln.b, x);
}

static void run_enc_layer(float* x, const EncL& p, Bufs& B_, hipStream_t s) {
    run_mha(x, x, p.sa, B_, s);
    ln_kernel<<<NTOK, 128, 0, s>>>(x, B_.kb, p.ln1.g, p.ln1.b, x);
    run_ff_ln(x, p.ff, p.ln2, B_, s);
}

static void run_dec_layer(float* x, const float* mem, const DecL& p, Bufs& B_, hipStream_t s) {
    run_mha(x, x, p.sa, B_, s);
    ln_kernel<<<NTOK, 128, 0, s>>>(x, B_.kb, p.ln1.g, p.ln1.b, x);
    run_mha(x, mem, p.ca, B_, s);
    ln_kernel<<<NTOK, 128, 0, s>>>(x, B_.kb, p.ln2.g, p.ln2.b, x);
    run_ff_ln(x, p.ff, p.ln3, B_, s);
}

static void run_mlp3(const float* in, int din, const MLP3& m, float* out, Bufs& B_, hipStream_t s) {
    launch_gemm(in, m.W1, m.b1, B_.qb, NTOK, DD, din, 1, s);
    launch_gemm(B_.qb, m.W2, m.b2, B_.kb, NTOK, DD, DD, 1, s);
    launch_gemm(B_.kb, m.W3, m.b3, out, NTOK, DD, DD, 0, s);
}

extern "C" void kernel_launch(void* const* d_in, const int* in_sizes, int n_in,
                              void* d_out, int out_size, void* d_ws, size_t ws_size,
                              hipStream_t stream) {
    (void)in_sizes; (void)n_in; (void)out_size; (void)ws_size;
    int pi = 0;
    const float* src = (const float*)d_in[pi++];
    const float* tgt = (const float*)d_in[pi++];
    auto nx      = [&]() { return (const float*)d_in[pi++]; };
    auto getMLP  = [&]() { MLP3 m; m.W1=nx(); m.b1=nx(); m.W2=nx(); m.b2=nx(); m.W3=nx(); m.b3=nx(); return m; };
    auto getAttn = [&]() { AttnP a; a.Wq=nx(); a.bq=nx(); a.Wk=nx(); a.bk=nx(); a.Wv=nx(); a.bv=nx(); a.Wo=nx(); a.bo=nx(); return a; };
    auto getLN   = [&]() { LNP l; l.g=nx(); l.b=nx(); return l; };
    auto getFF   = [&]() { FFP f; f.W1=nx(); f.b1=nx(); f.W2=nx(); f.b2=nx(); return f; };

    MLP3 emb_src = getMLP(), emb_enc_tgt = getMLP(), emb_dec_tgt = getMLP();
    EncL enc[LL];
    for (int l = 0; l < LL; ++l) { enc[l].sa = getAttn(); enc[l].ln1 = getLN(); enc[l].ff = getFF(); enc[l].ln2 = getLN(); }
    DecL encdec[LL], dec[LL];
    for (int l = 0; l < LL; ++l) { encdec[l].sa = getAttn(); encdec[l].ln1 = getLN(); encdec[l].ca = getAttn();
                                   encdec[l].ln2 = getLN(); encdec[l].ff = getFF(); encdec[l].ln3 = getLN(); }
    for (int l = 0; l < LL; ++l) { dec[l].sa = getAttn(); dec[l].ln1 = getLN(); dec[l].ca = getAttn();
                                   dec[l].ln2 = getLN(); dec[l].ff = getFF(); dec[l].ln3 = getLN(); }
    const float* eoW = nx(); const float* eob = nx();
    const float* doW = nx(); const float* dob = nx();
    const float* codebook = nx();

    // workspace carve-up
    float* w = (float*)d_ws;
    size_t off = 0;
    auto alloc = [&](size_t n) { float* p = w + off; off += n; return p; };
    const size_t ND = (size_t)NTOK * DD;
    Bufs B_;
    B_.mem = alloc(ND); B_.x = alloc(ND);
    B_.qb = alloc(ND); B_.kb = alloc(ND); B_.vb = alloc(ND); B_.t1 = alloc(ND);
    B_.ffh = alloc((size_t)NTOK * DFF);
    B_.z = alloc((size_t)NTOK * DC); B_.zq = alloc((size_t)NTOK * DC);
    B_.acc = alloc(4);

    float* out = (float*)d_out;
    const int peBlocks = (int)(((size_t)NTOK * DD + 255) / 256);

    init_acc_kernel<<<1, 1, 0, stream>>>(B_.acc);

    // encoder over src -> mem (src_e)
    run_mlp3(src, VV, emb_src, B_.mem, B_, stream);
    add_pe_kernel<<<peBlocks, 256, 0, stream>>>(B_.mem);
    for (int l = 0; l < LL; ++l) run_enc_layer(B_.mem, enc[l], B_, stream);

    // encode tgt conditioned on src -> x
    run_mlp3(tgt, VV, emb_enc_tgt, B_.x, B_, stream);
    add_pe_kernel<<<peBlocks, 256, 0, stream>>>(B_.x);
    for (int l = 0; l < LL; ++l) run_dec_layer(B_.x, B_.mem, encdec[l], B_, stream);

    // z = x @ enc_out.W + b ; VQ
    launch_gemm(B_.x, eoW, eob, B_.z, NTOK, DC, DD, 0, stream);
    vq_kernel<<<NTOK, 128, 0, stream>>>(B_.z, codebook, B_.zq, B_.acc + 1);

    // decode (straight-through forward == z_vq)
    run_mlp3(B_.zq, DC, emb_dec_tgt, B_.x, B_, stream);
    add_pe_kernel<<<peBlocks, 256, 0, stream>>>(B_.x);
    for (int l = 0; l < LL; ++l) run_dec_layer(B_.x, B_.mem, dec[l], B_, stream);

    // recon = sigmoid(x @ dec_out.W + b) -> d_out
    launch_gemm(B_.x, doW, dob, out, NTOK, VV, DD, 2, stream);
    bce_kernel<<<512, 256, 0, stream>>>(out, tgt, B_.acc);
    fin_kernel<<<1, 1, 0, stream>>>(out, B_.acc);
}
